// CustomMultiheadAttention_71442486002222
// MI455X (gfx1250) — matrix operand order repaired and checked by reading, not compiled, not run on hardware
//
#include <hip/hip_runtime.h>
#include <hip/hip_bf16.h>

// ---------------------------------------------------------------------------
// Problem constants (from the reference)
// ---------------------------------------------------------------------------
#define EMBED  1024
#define NHEAD  16
#define HDIM   64
#define SEQ    2048
#define BATCH  2
#define NQKV   3072              // 3*EMBED
#define MROWS  (BATCH * SEQ)     // 4096

typedef __bf16 bf16_t;
typedef __attribute__((ext_vector_type(16))) __bf16 v16bf;
typedef __attribute__((ext_vector_type(8)))  __bf16 v8bf;
typedef __attribute__((ext_vector_type(4)))  __bf16 v4bf;
typedef __attribute__((ext_vector_type(8)))  float  v8f;
typedef __attribute__((ext_vector_type(4)))  float  v4f;
typedef __attribute__((ext_vector_type(4)))  int    v4i;

#define WMMA_BF16(a, b, c) \
  __builtin_amdgcn_wmma_f32_16x16x32_bf16(false, (a), false, (b), (short)0, (c), false, false)

// ---------------------------------------------------------------------------
// gfx1250 async global->LDS copy (ASYNCcnt path), with graceful fallback.
// Probe-learned signature: (global int4*, lds int4*, imm offset, imm cpol)
// ---------------------------------------------------------------------------
#if defined(__has_builtin)
#  if __has_builtin(__builtin_amdgcn_global_load_async_to_lds_b128)
#    define HAVE_ASYNC_LDS 1
#  endif
#endif
#ifndef HAVE_ASYNC_LDS
#  define HAVE_ASYNC_LDS 0
#endif

#if HAVE_ASYNC_LDS
static __device__ __forceinline__ void async_cp16(const bf16_t* g, bf16_t* l) {
  __builtin_amdgcn_global_load_async_to_lds_b128(
      (__attribute__((address_space(1))) v4i*)(g),
      (__attribute__((address_space(3))) v4i*)(l), 0, 0);
}
static __device__ __forceinline__ void wait_async0() {
#  if __has_builtin(__builtin_amdgcn_s_wait_asynccnt)
  __builtin_amdgcn_s_wait_asynccnt(0);
#  else
  asm volatile("s_wait_asynccnt 0" ::: "memory");
#  endif
}
#endif

// ---------------------------------------------------------------------------
// WMMA fragment loaders (wave32; this code's contiguous k order k = 16*half + s, same for A and B)
// A 16x32 bf16: lane<16 -> row=lane, elems 0..15 = K[0..15]
//               lane>=16 -> row=lane-16, elems 0..15 = K[16..31]
// ---------------------------------------------------------------------------
static __device__ __forceinline__ v16bf load_a_frag(const bf16_t* base, int stride, int lane) {
  const int half = (lane >> 4) & 1;
  const int r    = lane & 15;
  const bf16_t* p = base + (size_t)r * stride + half * 16;
  union { v16bf v; v8bf h[2]; } u;
  u.h[0] = *(const v8bf*)(p);
  u.h[1] = *(const v8bf*)(p + 8);
  return u.v;
}

// B 32x16 bf16 (stored column-major == rows of W): lane<16 -> col=lane, K[0..15]
//                                                  lane>=16 -> col=lane-16, K[16..31]
static __device__ __forceinline__ v16bf load_b_frag(const bf16_t* base, int stride, int lane) {
  const int half = (lane >> 4) & 1;
  const int c    = lane & 15;
  const bf16_t* p = base + (size_t)c * stride + half * 16;
  union { v16bf v; v8bf h[2]; } u;
  u.h[0] = *(const v8bf*)(p);
  u.h[1] = *(const v8bf*)(p + 8);
  return u.v;
}

// ---------------------------------------------------------------------------
// fp32 -> bf16 conversion (vector x4)
// ---------------------------------------------------------------------------
__global__ void __launch_bounds__(256)
k_f32_to_bf16(const float* __restrict__ src, bf16_t* __restrict__ dst, int n) {
  int i = (blockIdx.x * 256 + threadIdx.x) * 4;
  if (i + 3 < n) {
    v4f f = *(const v4f*)(src + i);
    v4bf o;
    o.x = (bf16_t)f.x; o.y = (bf16_t)f.y; o.z = (bf16_t)f.z; o.w = (bf16_t)f.w;
    *(v4bf*)(dst + i) = o;
  }
}

// ---------------------------------------------------------------------------
// GEMM: C[M,N] = A[M,K](bf16,row) x W[N,K]^T(bf16,row) + bias[N]  (fp32 out)
// block = 256 thr = 8 waves in 4(M) x 2(N) grid; wave tile 32x64 (2x4 WMMA)
// block tile = 128 x 128. Per K-step: 12 b128 loads feed 8 WMMAs.
// ---------------------------------------------------------------------------
__global__ void __launch_bounds__(256)
k_gemm_bf16(const bf16_t* __restrict__ A, const bf16_t* __restrict__ W,
            const float* __restrict__ bias, float* __restrict__ C,
            int M, int N, int K) {
  const int lane = threadIdx.x & 31;
  const int w    = threadIdx.x >> 5;
  const int m0   = blockIdx.x * 128 + (w & 3) * 32;
  const int n0   = blockIdx.y * 128 + (w >> 2) * 64;

  v8f acc[2][4] = {};
  const bf16_t* Ar0 = A + (size_t)m0 * K;
  const bf16_t* Ar1 = A + (size_t)(m0 + 16) * K;
  const bf16_t* Wr  = W + (size_t)n0 * K;

  for (int k0 = 0; k0 < K; k0 += 32) {
    v16bf a0 = load_a_frag(Ar0 + k0, K, lane);
    v16bf a1 = load_a_frag(Ar1 + k0, K, lane);
    v16bf b[4];
#pragma unroll
    for (int ct = 0; ct < 4; ++ct)
      b[ct] = load_b_frag(Wr + (size_t)(ct * 16) * K + k0, K, lane);
#pragma unroll
    for (int ct = 0; ct < 4; ++ct) {
      acc[0][ct] = WMMA_BF16(a0, b[ct], acc[0][ct]);
      acc[1][ct] = WMMA_BF16(a1, b[ct], acc[1][ct]);
    }
  }

  const int half = lane >> 4, cl = lane & 15;
#pragma unroll
  for (int ct = 0; ct < 4; ++ct) {
    const int   n  = n0 + ct * 16 + cl;
    const float bs = bias[n];
#pragma unroll
    for (int j = 0; j < 8; ++j) {
      const int r0 = m0 + half * 8 + j;
      C[(size_t)r0 * N + n]        = acc[0][ct][j] + bs;
      C[(size_t)(r0 + 16) * N + n] = acc[1][ct][j] + bs;
    }
  }
}

// ---------------------------------------------------------------------------
// RoPE (fp32) + scatter to head-major bf16 tensors.
// Qh,Kh: [B*H, S, D]   Vt: [B*H, D, S] (transposed so PV B-frags are contiguous)
// One 32-lane wave handles one (b,h,s); lane t handles dim pair (2t, 2t+1).
// ---------------------------------------------------------------------------
__global__ void __launch_bounds__(256)
k_rope_scatter(const float* __restrict__ qkv,   // [B*S, 3*E]
               bf16_t* __restrict__ Qh, bf16_t* __restrict__ Kh,
               bf16_t* __restrict__ Vt) {
  const int gid = blockIdx.x * 8 + (threadIdx.x >> 5);   // (bh, s) flattened
  const int t   = threadIdx.x & 31;
  const int s   = gid & (SEQ - 1);
  const int bh  = gid >> 11;
  const int h   = bh & (NHEAD - 1);
  const int b   = bh >> 4;

  const int dc = 2 * t;
  const float inv_freq = __powf(10000.0f, -((float)dc) / (float)HDIM);
  const float ang = (float)s * inv_freq;
  const float c  = __cosf(ang);
  const float sn = __sinf(ang);

  const size_t row = (size_t)(b * SEQ + s) * NQKV;
  const float qe = qkv[row + h * HDIM + dc];
  const float qo = qkv[row + h * HDIM + dc + 1];
  const float ke = qkv[row + EMBED + h * HDIM + dc];
  const float ko = qkv[row + EMBED + h * HDIM + dc + 1];
  const float ve = qkv[row + 2 * EMBED + h * HDIM + dc];
  const float vo = qkv[row + 2 * EMBED + h * HDIM + dc + 1];

  const size_t qkbase = ((size_t)bh * SEQ + s) * HDIM + dc;
  Qh[qkbase]     = (bf16_t)(qe * c - qo * sn);
  Qh[qkbase + 1] = (bf16_t)(qe * sn + qo * c);
  Kh[qkbase]     = (bf16_t)(ke * c - ko * sn);
  Kh[qkbase + 1] = (bf16_t)(ke * sn + ko * c);

  const size_t vtb = ((size_t)bh * HDIM + dc) * SEQ + s;
  Vt[vtb]       = (bf16_t)ve;
  Vt[vtb + SEQ] = (bf16_t)vo;
}

// ---------------------------------------------------------------------------
// Flash attention. Block = 128 thr (4 waves). Each block: one (b,h) and a
// 64-row Q block. K/V 64x64 tiles are staged in LDS cooperatively (async
// global->LDS on gfx1250, ASYNCcnt + barrier), shared by all 4 waves
// (4x traffic reduction). Each wave owns 16 q-rows: full score rows ->
// in-wave online softmax using 16-lane half-wave shuffle reductions
// (wave32 C-frag layout puts row m's 16 values in one half). P tile is
// wave-private (each wave writes and reads only its own 16 LDS rows), so no
// barrier is needed around it — LDS ops are in-order within a wave.
// ---------------------------------------------------------------------------
__global__ void __launch_bounds__(128)
k_attention(const bf16_t* __restrict__ Qh, const bf16_t* __restrict__ Kh,
            const bf16_t* __restrict__ Vt, bf16_t* __restrict__ attn /*[B*S,E]*/) {
  __shared__ bf16_t Kt[64][72];     // key-block tile: rows = key, cols = d
  __shared__ bf16_t Vtile[64][72];  // value tile: rows = d, cols = key
  __shared__ bf16_t Pt[64][72];     // probs tile (wave-private row bands)

  const int tid  = threadIdx.x;
  const int lane = tid & 31;
  const int wv   = tid >> 5;                    // 0..3
  const int bh   = blockIdx.y;                  // 0..B*H-1
  const int h    = bh & (NHEAD - 1);
  const int b    = bh >> 4;
  const int q0   = blockIdx.x * 64;
  const int qr   = q0 + wv * 16;                // this wave's q-row base

  const bf16_t* Qbase = Qh + ((size_t)bh * SEQ + qr) * HDIM;
  const bf16_t* Kbase = Kh + (size_t)bh * SEQ * HDIM;
  const bf16_t* Vbase = Vt + (size_t)bh * HDIM * SEQ;

  const v16bf qf0 = load_a_frag(Qbase, HDIM, lane);        // d = 0..31
  const v16bf qf1 = load_a_frag(Qbase + 32, HDIM, lane);   // d = 32..63

  v8f o0 = {}, o1 = {}, o2 = {}, o3 = {};
  float mrow[8], lrow[8];
#pragma unroll
  for (int j = 0; j < 8; ++j) { mrow[j] = -3.0e38f; lrow[j] = 0.0f; }

  const int half = lane >> 4, cl = lane & 15;
  const float scale = 0.125f;                   // 1/sqrt(64)

  for (int kb = 0; kb < SEQ; kb += 64) {
    // ---- stage K and V 64x64 tiles into LDS (all 128 threads) ----------
    __syncthreads();                 // previous iteration's tile reads done
#pragma unroll
    for (int i = 0; i < 4; ++i) {
      const int chunk = i * 128 + tid;          // 0..511 (16B chunks)
      const int row   = chunk >> 3;             // 0..63
      const int c8    = (chunk & 7) * 8;        // element column
      const bf16_t* gk = Kbase + (size_t)(kb + row) * HDIM + c8;
      const bf16_t* gv = Vbase + (size_t)row * SEQ + kb + c8;
#if HAVE_ASYNC_LDS
      async_cp16(gk, &Kt[row][c8]);
      async_cp16(gv, &Vtile[row][c8]);
#else
      *(v8bf*)&Kt[row][c8]    = *(const v8bf*)gk;
      *(v8bf*)&Vtile[row][c8] = *(const v8bf*)gv;
#endif
    }
#if HAVE_ASYNC_LDS
    wait_async0();
#endif
    __syncthreads();                 // tiles visible to all waves

    // ---- scores S = Q * K^T for 4 column tiles of 16 keys --------------
    v8f sc[4];
#pragma unroll
    for (int ct = 0; ct < 4; ++ct) {
      v8f c = {};
      c = WMMA_BF16(qf0, load_b_frag(&Kt[ct * 16][0], 72, lane), c);
      c = WMMA_BF16(qf1, load_b_frag(&Kt[ct * 16][32], 72, lane), c);
      sc[ct] = c * scale;
    }
    // ---- online softmax: row max / sum via 16-lane half reductions -----
    float mnew[8];
#pragma unroll
    for (int j = 0; j < 8; ++j)
      mnew[j] = fmaxf(fmaxf(sc[0][j], sc[1][j]), fmaxf(sc[2][j], sc[3][j]));
#pragma unroll
    for (int off = 1; off < 16; off <<= 1)
#pragma unroll
      for (int j = 0; j < 8; ++j)
        mnew[j] = fmaxf(mnew[j], __shfl_xor(mnew[j], off, 32));
#pragma unroll
    for (int j = 0; j < 8; ++j) mnew[j] = fmaxf(mnew[j], mrow[j]);

    float alpha[8], rsum[8];
#pragma unroll
    for (int j = 0; j < 8; ++j) { alpha[j] = __expf(mrow[j] - mnew[j]); rsum[j] = 0.0f; }

    v8f p[4];
#pragma unroll
    for (int ct = 0; ct < 4; ++ct)
#pragma unroll
      for (int j = 0; j < 8; ++j) {
        float e = __expf(sc[ct][j] - mnew[j]);
        p[ct][j] = e;
        rsum[j] += e;
      }
#pragma unroll
    for (int off = 1; off < 16; off <<= 1)
#pragma unroll
      for (int j = 0; j < 8; ++j) rsum[j] += __shfl_xor(rsum[j], off, 32);

#pragma unroll
    for (int j = 0; j < 8; ++j) {
      lrow[j] = lrow[j] * alpha[j] + rsum[j];
      mrow[j] = mnew[j];
      o0[j] *= alpha[j]; o1[j] *= alpha[j]; o2[j] *= alpha[j]; o3[j] *= alpha[j];
    }

    // ---- C-layout P -> LDS (wave-private rows; in-wave DS ordering) ----
#pragma unroll
    for (int ct = 0; ct < 4; ++ct) {
      const int col = ct * 16 + cl;
#pragma unroll
      for (int j = 0; j < 8; ++j)
        Pt[wv * 16 + half * 8 + j][col] = (bf16_t)p[ct][j];
    }

    // ---- O += P * V  (P A-frags and V B-frags from LDS) ----------------
#pragma unroll
    for (int ki = 0; ki < 2; ++ki) {
      v16bf pf = load_a_frag(&Pt[wv * 16][ki * 32], 72, lane);
      o0 = WMMA_BF16(pf, load_b_frag(&Vtile[0][ki * 32], 72, lane), o0);
      o1 = WMMA_BF16(pf, load_b_frag(&Vtile[16][ki * 32], 72, lane), o1);
      o2 = WMMA_BF16(pf, load_b_frag(&Vtile[32][ki * 32], 72, lane), o2);
      o3 = WMMA_BF16(pf, load_b_frag(&Vtile[48][ki * 32], 72, lane), o3);
    }
  }

  // ---- finalize: O /= l, write bf16 attn [B*S, E] ----------------------
#pragma unroll
  for (int j = 0; j < 8; ++j) {
    const float inv = 1.0f / lrow[j];
    const size_t row = (size_t)(b * SEQ + qr + half * 8 + j) * EMBED + h * HDIM + cl;
    attn[row +  0] = (bf16_t)(o0[j] * inv);
    attn[row + 16] = (bf16_t)(o1[j] * inv);
    attn[row + 32] = (bf16_t)(o2[j] * inv);
    attn[row + 48] = (bf16_t)(o3[j] * inv);
  }
}

// ---------------------------------------------------------------------------
// Launch
// ---------------------------------------------------------------------------
extern "C" void kernel_launch(void* const* d_in, const int* in_sizes, int n_in,
                              void* d_out, int out_size, void* d_ws, size_t ws_size,
                              hipStream_t stream) {
  (void)in_sizes; (void)n_in; (void)out_size; (void)ws_size;

  const float* x     = (const float*)d_in[0];   // query (== key == value)
  const float* w_in  = (const float*)d_in[3];   // [3E, E]
  const float* b_in  = (const float*)d_in[4];   // [3E]
  const float* w_out = (const float*)d_in[5];   // [E, E]
  const float* b_out = (const float*)d_in[6];   // [E]
  float* out = (float*)d_out;

  // workspace layout (all offsets 256B-aligned)
  char* ws = (char*)d_ws;
  const size_t SZ_XB   = (size_t)MROWS * EMBED * 2;        //  8 MB bf16 x
  const size_t SZ_WIB  = (size_t)NQKV * EMBED * 2;         //  6 MB bf16 in_proj_w
  const size_t SZ_WOB  = (size_t)EMBED * EMBED * 2;        //  2 MB bf16 out_proj_w
  const size_t SZ_QKV  = (size_t)MROWS * NQKV * 4;         // 48 MB f32 qkv
  const size_t SZ_HEAD = (size_t)BATCH * NHEAD * SEQ * HDIM * 2; // 8 MB each

  bf16_t* xb   = (bf16_t*)(ws);
  bf16_t* wib  = (bf16_t*)(ws + SZ_XB);
  bf16_t* wob  = (bf16_t*)(ws + SZ_XB + SZ_WIB);
  float*  qkv  = (float*) (ws + SZ_XB + SZ_WIB + SZ_WOB);
  bf16_t* Qh   = (bf16_t*)(ws + SZ_XB + SZ_WIB + SZ_WOB + SZ_QKV);
  bf16_t* Kh   = (bf16_t*)(ws + SZ_XB + SZ_WIB + SZ_WOB + SZ_QKV + SZ_HEAD);
  bf16_t* Vt   = (bf16_t*)(ws + SZ_XB + SZ_WIB + SZ_WOB + SZ_QKV + 2 * SZ_HEAD);
  bf16_t* attn = (bf16_t*)(ws + SZ_XB + SZ_WIB + SZ_WOB + SZ_QKV + 3 * SZ_HEAD);

  // 1) fp32 -> bf16 converts
  {
    int n;
    n = MROWS * EMBED;  k_f32_to_bf16<<<n / 1024, 256, 0, stream>>>(x, xb, n);
    n = NQKV * EMBED;   k_f32_to_bf16<<<n / 1024, 256, 0, stream>>>(w_in, wib, n);
    n = EMBED * EMBED;  k_f32_to_bf16<<<n / 1024, 256, 0, stream>>>(w_out, wob, n);
  }

  // 2) QKV projection: [4096,1024] x [3072,1024]^T -> f32 [4096,3072]
  k_gemm_bf16<<<dim3(MROWS / 128, NQKV / 128), 256, 0, stream>>>(
      xb, wib, b_in, qkv, MROWS, NQKV, EMBED);

  // 3) RoPE + scatter to head-major bf16 (V transposed)
  k_rope_scatter<<<(BATCH * NHEAD * SEQ) / 8, 256, 0, stream>>>(qkv, Qh, Kh, Vt);

  // 4) attention (flash, WMMA, async-LDS-staged K/V) -> bf16 attn [4096,1024]
  k_attention<<<dim3(SEQ / 64, BATCH * NHEAD), 128, 0, stream>>>(Qh, Kh, Vt, attn);

  // 5) output projection: [4096,1024] x [1024,1024]^T + bias -> f32 out
  k_gemm_bf16<<<dim3(MROWS / 128, EMBED / 128), 256, 0, stream>>>(
      attn, wob, b_out, out, MROWS, EMBED, EMBED);
}
